// MultiHeadAttention_42520176230575
// MI455X (gfx1250) — matrix operand order repaired and checked by reading, not compiled, not run on hardware
//
#include <hip/hip_runtime.h>
#include <math.h>

#define DEV __device__ __forceinline__

typedef __attribute__((ext_vector_type(16))) __bf16 v16bf;
typedef __attribute__((ext_vector_type(8)))  __bf16 v8bf;
typedef __attribute__((ext_vector_type(8)))  float  v8f;

constexpr int cB = 2, cS = 2048, cD = 1024, cH = 16, cHD = 64;

// ---------- hardware fp32 -> bf16 (RNE) via vector convert ----------
DEV v8bf cvt8(v8f x) { return __builtin_convertvector(x, v8bf); }

DEV v16bf cat16(v8bf lo, v8bf hi) {
  return __builtin_shufflevector(lo, hi, 0, 1, 2, 3, 4, 5, 6, 7,
                                 8, 9, 10, 11, 12, 13, 14, 15);
}

// ---------- WMMA fragment loaders (ISA 7.12.2 layouts, wave32) ----------
// A 16x32 bf16 from row-major bf16 [*, lda]: lane half h elems 0..7 -> K=k0+8h..,
// elems 8..15 -> K=k0+16+8h..  (two contiguous 16B loads)
DEV v16bf load_a_bf(const __bf16* __restrict__ A, int lda, int m0, int k0, int lane) {
  int m = lane & 15, half = lane >> 4;
  const __bf16* p = A + (size_t)(m0 + m) * lda + k0 + half * 8;
  v8bf lo = *(const v8bf*)p;
  v8bf hi = *(const v8bf*)(p + 16);
  return cat16(lo, hi);
}

// A 16x32 from row-major fp32, non-temporal (streaming operand) + hw cvt
DEV v16bf load_a_f32_nt(const float* __restrict__ A, int lda, int m0, int k0, int lane) {
  int m = lane & 15, half = lane >> 4;
  const float* p = A + (size_t)(m0 + m) * lda + k0 + half * 8;
  v8f lo = __builtin_nontemporal_load((const v8f*)p);
  v8f hi = __builtin_nontemporal_load((const v8f*)(p + 16));
  return cat16(cvt8(lo), cvt8(hi));
}

// B 32x16 where B(k,n) = W[n][k] (B = W^T), W row-major bf16 [N,K]:
// lane -> N, elems 0..7 -> K = k0 + 8*half.., elems 8..15 -> K = k0 + 16 + 8*half..  (two 16B loads)
DEV v16bf load_bT_bf(const __bf16* __restrict__ W, int ldw, int n0, int k0, int lane) {
  int n = lane & 15, half = lane >> 4;
  const __bf16* p = W + (size_t)(n0 + n) * ldw + k0 + half * 8;
  v8bf lo = *(const v8bf*)p;
  v8bf hi = *(const v8bf*)(p + 16);
  return cat16(lo, hi);
}

DEV v8f wmma_bf16(v16bf a, v16bf b, v8f c) {
  return __builtin_amdgcn_wmma_f32_16x16x32_bf16(false, a, false, b, (short)0, c,
                                                 false, false);
}

// =======================================================================
// Kernel 0: elementwise fp32 -> bf16 (one-shot operand conversion)
// =======================================================================
__global__ __launch_bounds__(256) void cvt_f32_bf16(const float* __restrict__ in,
                                                    __bf16* __restrict__ out, int n) {
  int i = (blockIdx.x * 256 + threadIdx.x) * 8;
  if (i + 8 > n) return;
  v8f x = *(const v8f*)(in + i);
  *(v8bf*)(out + i) = cvt8(x);
}

// =======================================================================
// Kernel 1: C = A[M,K](bf16) @ W[N,K](bf16)^T
// MODE 0: f32 flat [M,N], NT stores   (final output projection)
// MODE 1: bf16 split-head [B,H,S,HD]  (q,k heads)
// MODE 2: bf16 split-head transposed [B,H,HD,S] (v heads; contiguous stores)
// 8 waves as 2(m) x 4(n); wave tile 32x64; block tile 64x256.
// =======================================================================
template <int MODE>
__global__ __launch_bounds__(256) void gemm_bf16(const __bf16* __restrict__ A,
                                                 const __bf16* __restrict__ W,
                                                 void* __restrict__ outp,
                                                 int M, int N, int K) {
  int lane = threadIdx.x & 31, wid = threadIdx.x >> 5;
  int wm = wid >> 2, wn = wid & 3;
  int m0 = blockIdx.x * 64 + wm * 32;
  int n0 = blockIdx.y * 256 + wn * 64;

  v8f acc[2][4] = {};
  for (int k0 = 0; k0 < K; k0 += 32) {
    v16bf a0 = load_a_bf(A, K, m0, k0, lane);
    v16bf a1 = load_a_bf(A, K, m0 + 16, k0, lane);
    v16bf b[4];
#pragma unroll
    for (int j = 0; j < 4; ++j) b[j] = load_bT_bf(W, K, n0 + j * 16, k0, lane);
#pragma unroll
    for (int j = 0; j < 4; ++j) {
      acc[0][j] = wmma_bf16(a0, b[j], acc[0][j]);
      acc[1][j] = wmma_bf16(a1, b[j], acc[1][j]);
    }
  }

  int half = lane >> 4, nl = lane & 15;
#pragma unroll
  for (int ti = 0; ti < 2; ++ti)
#pragma unroll
    for (int tj = 0; tj < 4; ++tj) {
      v8f c = acc[ti][tj];
      if (MODE == 0) {
        float* out = (float*)outp;
#pragma unroll
        for (int r = 0; r < 8; ++r) {
          int m = m0 + ti * 16 + r + 8 * half;
          int n = n0 + tj * 16 + nl;
          __builtin_nontemporal_store(c[r], &out[(size_t)m * N + n]);
        }
      } else if (MODE == 1) {
        __bf16* out = (__bf16*)outp;
        v8bf cv = cvt8(c);
#pragma unroll
        for (int r = 0; r < 8; ++r) {
          int m = m0 + ti * 16 + r + 8 * half;
          int n = n0 + tj * 16 + nl;
          int b = m / cS, s = m % cS, h = n / cHD, d = n % cHD;
          out[(((size_t)b * cH + h) * cS + s) * cHD + d] = cv[r];
        }
      } else {
        __bf16* out = (__bf16*)outp;
        v8bf cv = cvt8(c);
        int mbase = m0 + ti * 16 + 8 * half;  // 8-aligned run of s
        int n = n0 + tj * 16 + nl;
        int b = mbase / cS, s = mbase % cS, h = n / cHD, d = n % cHD;
        *(v8bf*)(out + (((size_t)b * cH + h) * cHD + d) * cS + s) = cv;
      }
    }
}

// =======================================================================
// Kernel 2: per (b,h) raw scores = q k^T / 8 (bf16 in, f32 NT-stored to d_out)
// wave tile 32x64; block tile 64x256; grid.z = b*H+h
// =======================================================================
__global__ __launch_bounds__(256) void attn_scores(const __bf16* __restrict__ q,
                                                   const __bf16* __restrict__ k,
                                                   float* __restrict__ w) {
  int bh = blockIdx.z;
  const __bf16* qp = q + (size_t)bh * cS * cHD;
  const __bf16* kp = k + (size_t)bh * cS * cHD;
  float* wp = w + (size_t)bh * cS * cS;

  int lane = threadIdx.x & 31, wid = threadIdx.x >> 5;
  int wm = wid >> 2, wn = wid & 3;
  int m0 = blockIdx.x * 64 + wm * 32;
  int n0 = blockIdx.y * 256 + wn * 64;

  v8f acc[2][4] = {};
#pragma unroll
  for (int k0 = 0; k0 < cHD; k0 += 32) {
    v16bf a0 = load_a_bf(qp, cHD, m0, k0, lane);
    v16bf a1 = load_a_bf(qp, cHD, m0 + 16, k0, lane);
    v16bf b[4];
#pragma unroll
    for (int j = 0; j < 4; ++j) b[j] = load_bT_bf(kp, cHD, n0 + j * 16, k0, lane);
#pragma unroll
    for (int j = 0; j < 4; ++j) {
      acc[0][j] = wmma_bf16(a0, b[j], acc[0][j]);
      acc[1][j] = wmma_bf16(a1, b[j], acc[1][j]);
    }
  }

  int half = lane >> 4, nl = lane & 15;
#pragma unroll
  for (int ti = 0; ti < 2; ++ti)
#pragma unroll
    for (int tj = 0; tj < 4; ++tj) {
      v8f c = acc[ti][tj];
#pragma unroll
      for (int r = 0; r < 8; ++r) {
        int m = m0 + ti * 16 + r + 8 * half;
        int n = n0 + tj * 16 + nl;
        __builtin_nontemporal_store(c[r] * 0.125f, &wp[(size_t)m * cS + n]);
      }
    }
}

// =======================================================================
// Kernel 3: in-place row softmax over 2048 columns.
// 1 block (8 waves) per row; 8 contiguous elems/thread; wave32 shuffle
// reductions + 8-entry LDS stage; NT load/store (row used once per pass).
// =======================================================================
__global__ __launch_bounds__(256) void softmax_rows(float* __restrict__ w) {
  __shared__ float redmax[8];
  __shared__ float redsum[8];
  int t = threadIdx.x, lane = t & 31, wid = t >> 5;
  float* p = w + (size_t)blockIdx.x * cS + t * 8;

  v8f x = __builtin_nontemporal_load((const v8f*)p);

  float m = x[0];
#pragma unroll
  for (int i = 1; i < 8; ++i) m = fmaxf(m, x[i]);
#pragma unroll
  for (int off = 16; off; off >>= 1) m = fmaxf(m, __shfl_xor(m, off, 32));
  if (lane == 0) redmax[wid] = m;
  __syncthreads();
#pragma unroll
  for (int i = 0; i < 8; ++i) m = fmaxf(m, redmax[i]);

  v8f e;
  float sum = 0.f;
#pragma unroll
  for (int i = 0; i < 8; ++i) {
    e[i] = __expf(x[i] - m);
    sum += e[i];
  }
#pragma unroll
  for (int off = 16; off; off >>= 1) sum += __shfl_xor(sum, off, 32);
  if (lane == 0) redsum[wid] = sum;
  __syncthreads();
  sum = 0.f;
#pragma unroll
  for (int i = 0; i < 8; ++i) sum += redsum[i];

  float inv = 1.0f / sum;
#pragma unroll
  for (int i = 0; i < 8; ++i) e[i] *= inv;
  __builtin_nontemporal_store(e, (v8f*)p);
}

// =======================================================================
// Kernel 4: per (b,h) out = weights(f32, NT + hw-cvt) @ v (bf16 [HD,S])
// -> concat bf16 [B,S,D]. 8 waves as 4(m) x 2(n); block tile 128x64.
// Streaming weight rows prefetched ~4KB ahead (global_prefetch_b8).
// =======================================================================
__global__ __launch_bounds__(256) void attn_v(const float* __restrict__ w,
                                              const __bf16* __restrict__ vT,
                                              __bf16* __restrict__ concat) {
  int bh = blockIdx.z, b = bh / cH, h = bh % cH;
  const float* wp = w + (size_t)bh * cS * cS;
  const __bf16* vp = vT + (size_t)bh * cS * cHD;  // [HD, S]

  int lane = threadIdx.x & 31, wid = threadIdx.x >> 5;
  int wm = wid >> 1, wn = wid & 1;
  int m0 = blockIdx.x * 128 + wm * 32;
  int n0 = wn * 32;

  // prefetch pointers: this lane's weight rows, 1024 floats (32 k-steps) ahead
  const float* pf0 = wp + (size_t)(m0 + (lane & 15)) * cS + (lane >> 4) * 8 + 1024;
  const float* pf1 = pf0 + (size_t)16 * cS;

  v8f acc[2][2] = {};
  for (int k0 = 0; k0 < cS; k0 += 32) {
    __builtin_prefetch(pf0, 0, 0);
    __builtin_prefetch(pf1, 0, 0);
    pf0 += 32;
    pf1 += 32;
    v16bf a0 = load_a_f32_nt(wp, cS, m0, k0, lane);
    v16bf a1 = load_a_f32_nt(wp, cS, m0 + 16, k0, lane);
    v16bf b0 = load_bT_bf(vp, cS, n0, k0, lane);       // B(k=j,n=d)=vT[d][j]
    v16bf b1 = load_bT_bf(vp, cS, n0 + 16, k0, lane);
    acc[0][0] = wmma_bf16(a0, b0, acc[0][0]);
    acc[0][1] = wmma_bf16(a0, b1, acc[0][1]);
    acc[1][0] = wmma_bf16(a1, b0, acc[1][0]);
    acc[1][1] = wmma_bf16(a1, b1, acc[1][1]);
  }

  int half = lane >> 4, nl = lane & 15;
#pragma unroll
  for (int ti = 0; ti < 2; ++ti)
#pragma unroll
    for (int tj = 0; tj < 2; ++tj) {
      v8bf cv = cvt8(acc[ti][tj]);
#pragma unroll
      for (int r = 0; r < 8; ++r) {
        int m = m0 + ti * 16 + r + 8 * half;  // sequence position
        int n = n0 + tj * 16 + nl;            // head dim
        concat[((size_t)b * cS + m) * cD + h * cHD + n] = cv[r];
      }
    }
}

// =======================================================================
// host-side launcher
// =======================================================================
extern "C" void kernel_launch(void* const* d_in, const int* in_sizes, int n_in,
                              void* d_out, int out_size, void* d_ws, size_t ws_size,
                              hipStream_t stream) {
  (void)in_sizes; (void)n_in; (void)out_size; (void)ws_size;

  const float* Q  = (const float*)d_in[0];
  const float* Km = (const float*)d_in[1];
  const float* V  = (const float*)d_in[2];
  const float* Wq = (const float*)d_in[3];
  const float* Wk = (const float*)d_in[4];
  const float* Wv = (const float*)d_in[5];
  const float* Wo = (const float*)d_in[6];

  const size_t big   = (size_t)cB * cS * cD;  // 4,194,304
  const size_t small = (size_t)cD * cD;       // 1,048,576

  __bf16* Qb     = (__bf16*)d_ws;
  __bf16* Kb     = Qb + big;
  __bf16* Vb     = Kb + big;
  __bf16* Wqb    = Vb + big;
  __bf16* Wkb    = Wqb + small;
  __bf16* Wvb    = Wkb + small;
  __bf16* Wob    = Wvb + small;
  __bf16* qh     = Wob + small;  // [B,H,S,HD] bf16
  __bf16* kh     = qh + big;     // [B,H,S,HD] bf16
  __bf16* vhT    = kh + big;     // [B,H,HD,S] bf16
  __bf16* concat = vhT + big;    // [B,S,D]    bf16   (total ~67 MB of ws)

  float* out_final = (float*)d_out;   // [B,S,D]
  float* out_w     = out_final + big; // [B,H,S,S]

  dim3 blk(256);

  // 0) one-shot fp32 -> bf16 operand conversion
  cvt_f32_bf16<<<dim3(big / 2048), blk, 0, stream>>>(Q,  Qb,  (int)big);
  cvt_f32_bf16<<<dim3(big / 2048), blk, 0, stream>>>(Km, Kb,  (int)big);
  cvt_f32_bf16<<<dim3(big / 2048), blk, 0, stream>>>(V,  Vb,  (int)big);
  cvt_f32_bf16<<<dim3(small / 2048), blk, 0, stream>>>(Wq, Wqb, (int)small);
  cvt_f32_bf16<<<dim3(small / 2048), blk, 0, stream>>>(Wk, Wkb, (int)small);
  cvt_f32_bf16<<<dim3(small / 2048), blk, 0, stream>>>(Wv, Wvb, (int)small);
  cvt_f32_bf16<<<dim3(small / 2048), blk, 0, stream>>>(Wo, Wob, (int)small);

  // 1) projections (q,k split-head; v split-head transposed)
  dim3 gproj(cB * cS / 64, cD / 256, 1);
  gemm_bf16<1><<<gproj, blk, 0, stream>>>(Qb, Wqb, qh,  cB * cS, cD, cD);
  gemm_bf16<1><<<gproj, blk, 0, stream>>>(Kb, Wkb, kh,  cB * cS, cD, cD);
  gemm_bf16<2><<<gproj, blk, 0, stream>>>(Vb, Wvb, vhT, cB * cS, cD, cD);

  // 2) raw scores -> weights region of d_out (NT stores)
  dim3 gsc(cS / 64, cS / 256, cB * cH);
  attn_scores<<<gsc, blk, 0, stream>>>(qh, kh, out_w);

  // 3) softmax in place (NT pass over 537 MB)
  softmax_rows<<<dim3(cB * cH * cS), blk, 0, stream>>>(out_w);

  // 4) weights @ v -> concat (bf16)
  dim3 gav(cS / 128, 1, cB * cH);
  attn_v<<<gav, blk, 0, stream>>>(out_w, vhT, concat);

  // 5) final output projection (f32 NT out)
  gemm_bf16<0><<<gproj, blk, 0, stream>>>(concat, Wob, out_final, cB * cS, cD, cD);
}
